// FiLM_Net_Graph_71975061946540
// MI455X (gfx1250) — compile-verified, matrix operand order checked by reading, not run on hardware
//
#include <hip/hip_runtime.h>
#include <hip/hip_bf16.h>
#include <stdint.h>

// ---------------------------------------------------------------------------
// CDNA5 (gfx1250) FiLM-GNN. Dense node transforms run on v_wmma_f32_16x16x32_bf16
// (wave32, fp32 accumulate). A/B operands are pre-swizzled into the ISA 7.12.2
// per-lane register layout; each GEMM wave owns a 32-row x 64-col macro-tile
// (8 independent WMMA accumulator chains, ~110 VGPRs -> no spills).
// ---------------------------------------------------------------------------

typedef __attribute__((ext_vector_type(16))) __bf16 v16bf;
typedef __attribute__((ext_vector_type(8)))  float  v8f;

#define RM 2   // row tiles (16 rows each) per wave

// ISA 7.12.2, 16-bit A/B 16x32 tile: lane L holds row/col (L&15); element e of
// the 16-element-per-lane vector maps to k = (e<8?0:16) + 8*(L>>4) + (e&7).
__device__ __forceinline__ int kOfElem(int lane, int e) {
    return ((e < 8) ? 0 : 16) + 8 * (lane >> 4) + (e & 7);
}

// ---------------------------------------------------------------------------
// Pack activations X[N,K] (f32 row-major) -> bf16 WMMA-A layout:
//   Apack[((rt*Kt + kt)*32 + lane)*16 + e]
// ---------------------------------------------------------------------------
__global__ void pack_a_kernel(const float* __restrict__ X, __bf16* __restrict__ A,
                              int Nrows, int K) {
    int Kt = K >> 5;
    int t = blockIdx.x * blockDim.x + threadIdx.x;
    int lane = t & 31;
    int tile = t >> 5;
    int RT = (Nrows + 15) >> 4;
    if (tile >= RT * Kt) return;
    int kt = tile % Kt, rt = tile / Kt;
    int m = (rt << 4) + (lane & 15);
    v16bf v;
#pragma unroll
    for (int e = 0; e < 16; ++e) {
        int k = (kt << 5) + kOfElem(lane, e);
        float xv = (m < Nrows) ? X[(size_t)m * K + k] : 0.0f;
        v[e] = (__bf16)xv;
    }
    *(v16bf*)(A + (((size_t)tile * 32) + lane) * 16) = v;
}

// ---------------------------------------------------------------------------
// Pack a dense weight W[K,Wout] (f32 row-major) -> bf16 WMMA-B layout:
//   Bswz[((ct*Kt + kt)*32 + lane)*16 + e]   (ct = 16-col tile)
// ---------------------------------------------------------------------------
__global__ void pack_b_dense(const float* __restrict__ W, __bf16* __restrict__ B,
                             int K, int Wout) {
    int Kt = K >> 5;
    int t = blockIdx.x * blockDim.x + threadIdx.x;
    int lane = t & 31;
    int tile = t >> 5;
    if (tile >= (Wout >> 4) * Kt) return;
    int kt = tile % Kt, ct = tile / Kt;
    int n = (ct << 4) + (lane & 15);
    v16bf v;
#pragma unroll
    for (int e = 0; e < 16; ++e) {
        int k = (kt << 5) + kOfElem(lane, e);
        v[e] = (__bf16)W[(size_t)k * Wout + n];
    }
    *(v16bf*)(B + (((size_t)tile * 32) + lane) * 16) = v;
}

// Concatenated FiLM layer weight matrix, 64 x 960 logical columns:
//   [0,64)   W_skip            [64,192)  F_skip (beta|gamma)
//   [192,448) W_rel r=0..3     [448,960) F_rel r=0..3 (beta|gamma each)
__device__ __forceinline__ float filmW(const float* Wsk, const float* Fsk,
                                       const float* Wrel, const float* Frel,
                                       int k, int c) {
    if (c < 64)  return Wsk[k * 64 + c];
    if (c < 192) return Fsk[k * 128 + (c - 64)];
    if (c < 448) { int r = (c - 192) >> 6; return Wrel[((size_t)(r * 64 + k)) * 64 + ((c - 192) & 63)]; }
    int r = (c - 448) >> 7;
    return Frel[((size_t)(r * 64 + k)) * 128 + ((c - 448) & 127)];
}

__global__ void pack_b_film(const float* __restrict__ Wsk, const float* __restrict__ Fsk,
                            const float* __restrict__ Wrel, const float* __restrict__ Frel,
                            __bf16* __restrict__ B) {
    const int Kt = 2, Wout = 960;
    int t = blockIdx.x * blockDim.x + threadIdx.x;
    int lane = t & 31;
    int tile = t >> 5;
    if (tile >= (Wout >> 4) * Kt) return;
    int kt = tile % Kt, ct = tile / Kt;
    int n = (ct << 4) + (lane & 15);
    v16bf v;
#pragma unroll
    for (int e = 0; e < 16; ++e) {
        int k = (kt << 5) + kOfElem(lane, e);
        v[e] = (__bf16)filmW(Wsk, Fsk, Wrel, Frel, k, n);
    }
    *(v16bf*)(B + (((size_t)tile * 32) + lane) * 16) = v;
}

// ---------------------------------------------------------------------------
// WMMA GEMM: Y[Nrows,Wout] = Apack @ Bswz (+bias). One wave per workgroup;
// wave computes a 32-row x 64-col macro-tile: 8 accumulators (64 VGPRs), so
// every B tile fetched feeds 2 WMMAs and 8 independent chains hide latency
// without spilling.
// C/D layout (ISA 7.12.2): acc[p] -> row rt*16 + p + 8*(lane>>4), col lane&15.
// ---------------------------------------------------------------------------
__device__ __forceinline__ void store_tile(float* __restrict__ Y, v8f acc,
                                           int mbase, int n, int Nrows, int Wout,
                                           float bb) {
#pragma unroll
    for (int p = 0; p < 8; ++p) {
        int m = mbase + p;
        if (m < Nrows) Y[(size_t)m * Wout + n] = acc[p] + bb;
    }
}

__global__ void __launch_bounds__(32)
gemm_wmma(const __bf16* __restrict__ A, const __bf16* __restrict__ B,
          float* __restrict__ Y, const float* __restrict__ bias,
          int Nrows, int Kt, int Wout, int RT) {
    int lane = threadIdx.x & 31;
    int rt0 = blockIdx.x * RM;   // first 16-row tile of this wave
    int cg  = blockIdx.y;        // 64-col group (4 WMMA col tiles)

    v8f acc[RM][4] = {};
    const size_t ctStride = (size_t)Kt * 512;  // elements between col tiles

    for (int kt = 0; kt < Kt; ++kt) {
        // Load RM A tiles (clamped for the ragged last block; masked at store).
        v16bf a[RM];
#pragma unroll
        for (int i = 0; i < RM; ++i) {
            int rti = rt0 + i; if (rti >= RT) rti = RT - 1;
            const __bf16* ap = A + (((size_t)rti * Kt + kt) * 32 + lane) * 16;
            a[i] = *(const v16bf*)ap;
            if (kt + 1 < Kt) __builtin_prefetch(ap + 512, 0, 0);  // global_prefetch_b8
        }
        const __bf16* bp = B + (((size_t)(cg * 4) * Kt + kt) * 32 + lane) * 16;
#pragma unroll
        for (int j = 0; j < 4; ++j) {
            v16bf b = *(const v16bf*)(bp + (size_t)j * ctStride);
#pragma unroll
            for (int i = 0; i < RM; ++i) {
                acc[i][j] = __builtin_amdgcn_wmma_f32_16x16x32_bf16(
                    false, a[i], false, b, (short)0, acc[i][j], false, false);
            }
        }
    }

    int half = lane >> 4, ncol = lane & 15;
    int n0 = cg * 64 + ncol;
    float bb[4];
#pragma unroll
    for (int j = 0; j < 4; ++j) bb[j] = bias ? bias[n0 + 16 * j] : 0.0f;
#pragma unroll
    for (int i = 0; i < RM; ++i) {
        int mbase = (rt0 + i) * 16 + 8 * half;
#pragma unroll
        for (int j = 0; j < 4; ++j)
            store_tile(Y, acc[i][j], mbase, n0 + 16 * j, Nrows, Wout, bb[j]);
    }
}

// ---------------------------------------------------------------------------
// Elementwise / edge-side kernels (float4-vectorized: b128 loads, less VALU/SALU)
// ---------------------------------------------------------------------------
__global__ void zero4_kernel(float4* __restrict__ p, size_t n4) {
    size_t t = (size_t)blockIdx.x * blockDim.x + threadIdx.x;
    if (t < n4) p[t] = make_float4(0.f, 0.f, 0.f, 0.f);
}

__global__ void edge_count_kernel(const int* __restrict__ dst, const int* __restrict__ et,
                                  float* __restrict__ cnt, int E, int Nn) {
    int e = blockIdx.x * blockDim.x + threadIdx.x;
    if (e >= E) return;
    atomicAdd(&cnt[(size_t)et[e] * Nn + dst[e]], 1.0f);
}

// msg = relu(gamma_r[r,dst]*h_rel[r,src] + beta_r[r,dst]); atomically summed
// per (r,dst). 16 lanes per edge, float4 per lane -> coalesced b128 gathers.
__global__ void edge_msg_kernel(const float* __restrict__ Y,
                                const int* __restrict__ src, const int* __restrict__ dst,
                                const int* __restrict__ et,
                                float* __restrict__ agg, int E, int Nn) {
    long long t = (long long)blockIdx.x * blockDim.x + threadIdx.x;
    int e = (int)(t >> 4);
    int q = (int)(t & 15);           // handles channels [4q, 4q+4)
    if (e >= E) return;
    int r = et[e], s = src[e], d = dst[e];
    const float4* yd = (const float4*)(Y + (size_t)d * 960 + 448 + r * 128);
    float4 beta  = yd[q];
    float4 gamma = yd[16 + q];
    float4 hs    = ((const float4*)(Y + (size_t)s * 960 + 192 + r * 64))[q];
    float4 m;
    m.x = fmaxf(fmaf(gamma.x, hs.x, beta.x), 0.0f);
    m.y = fmaxf(fmaf(gamma.y, hs.y, beta.y), 0.0f);
    m.z = fmaxf(fmaf(gamma.z, hs.z, beta.z), 0.0f);
    m.w = fmaxf(fmaf(gamma.w, hs.w, beta.w), 0.0f);
    float* ag = agg + (((size_t)r * Nn) + d) * 64 + q * 4;
    atomicAdd(ag + 0, m.x);
    atomicAdd(ag + 1, m.y);
    atomicAdd(ag + 2, m.z);
    atomicAdd(ag + 3, m.w);
}

// out = relu(gamma_s*skip + beta_s) + sum_r agg[r]/max(cnt[r],1); optional BN+ReLU.
__global__ void finalize_kernel(const float* __restrict__ Y, const float* __restrict__ agg,
                                const float* __restrict__ cnt,
                                const float* __restrict__ bn_g, const float* __restrict__ bn_b,
                                int useBn, float* __restrict__ Hout, int Nn) {
    int t = blockIdx.x * blockDim.x + threadIdx.x;
    if (t >= Nn * 16) return;
    int n = t >> 4, q = t & 15;
    const float4* yr = (const float4*)(Y + (size_t)n * 960);
    float4 skip  = yr[q];
    float4 beta  = yr[16 + q];
    float4 gamma = yr[32 + q];
    float4 val;
    val.x = fmaxf(fmaf(gamma.x, skip.x, beta.x), 0.0f);
    val.y = fmaxf(fmaf(gamma.y, skip.y, beta.y), 0.0f);
    val.z = fmaxf(fmaf(gamma.z, skip.z, beta.z), 0.0f);
    val.w = fmaxf(fmaf(gamma.w, skip.w, beta.w), 0.0f);
#pragma unroll
    for (int r = 0; r < 4; ++r) {
        float inv = 1.0f / fmaxf(cnt[(size_t)r * Nn + n], 1.0f);
        float4 a = ((const float4*)(agg + (((size_t)r * Nn) + n) * 64))[q];
        val.x = fmaf(a.x, inv, val.x);
        val.y = fmaf(a.y, inv, val.y);
        val.z = fmaf(a.z, inv, val.z);
        val.w = fmaf(a.w, inv, val.w);
    }
    if (useBn) {
        const float is = 1.0f / sqrtf(1.0f + 1e-5f);
        int h = q * 4;
        val.x = fmaxf(fmaf(val.x, bn_g[h + 0] * is, bn_b[h + 0]), 0.0f);
        val.y = fmaxf(fmaf(val.y, bn_g[h + 1] * is, bn_b[h + 1]), 0.0f);
        val.z = fmaxf(fmaf(val.z, bn_g[h + 2] * is, bn_b[h + 2]), 0.0f);
        val.w = fmaxf(fmaf(val.w, bn_g[h + 3] * is, bn_b[h + 3]), 0.0f);
    }
    ((float4*)Hout)[t] = val;
}

__global__ void pool_kernel(const float* __restrict__ Hf, const int* __restrict__ batch,
                            float* __restrict__ g, int Nn) {
    int t = blockIdx.x * blockDim.x + threadIdx.x;
    if (t >= Nn * 16) return;
    int n = t >> 4, q = t & 15;
    float4 v = ((const float4*)Hf)[t];
    float* gp = g + (size_t)batch[n] * 64 + q * 4;
    atomicAdd(gp + 0, v.x);
    atomicAdd(gp + 1, v.y);
    atomicAdd(gp + 2, v.z);
    atomicAdd(gp + 3, v.w);
}

// Head: relu(g @ lin_W + lin_b) @ clf_W + clf_b, G=128 rows. One workgroup.
__global__ void head_kernel(const float* __restrict__ g,
                            const float* __restrict__ linW, const float* __restrict__ linb,
                            const float* __restrict__ clfW, const float* __restrict__ clfb,
                            float* __restrict__ out) {
    __shared__ float sW[64 * 64];
    __shared__ float sC[64 * 10];
    int tid = threadIdx.x;
    for (int i = tid; i < 64 * 64; i += blockDim.x) sW[i] = linW[i];
    for (int i = tid; i < 64 * 10; i += blockDim.x) sC[i] = clfW[i];
    __syncthreads();
    if (tid >= 128) return;
    float gi[64];
#pragma unroll
    for (int k = 0; k < 64; ++k) gi[k] = g[(size_t)tid * 64 + k];
    float o[10];
#pragma unroll
    for (int q = 0; q < 10; ++q) o[q] = clfb[q];
    for (int j = 0; j < 64; ++j) {
        float tj = linb[j];
#pragma unroll
        for (int k = 0; k < 64; ++k) tj = fmaf(gi[k], sW[k * 64 + j], tj);
        tj = fmaxf(tj, 0.0f);
#pragma unroll
        for (int q = 0; q < 10; ++q) o[q] = fmaf(tj, sC[j * 10 + q], o[q]);
    }
#pragma unroll
    for (int q = 0; q < 10; ++q) out[(size_t)tid * 10 + q] = o[q];
}

// ---------------------------------------------------------------------------
extern "C" void kernel_launch(void* const* d_in, const int* in_sizes, int n_in,
                              void* d_out, int out_size, void* d_ws, size_t ws_size,
                              hipStream_t stream) {
    const float* x     = (const float*)d_in[0];
    const int*   eidx  = (const int*)d_in[1];
    const int*   etype = (const int*)d_in[2];
    const int*   batch = (const int*)d_in[3];
    const float* encW  = (const float*)d_in[4];
    const float* encB  = (const float*)d_in[5];
    const float* Wsk0  = (const float*)d_in[6];
    const float* Fsk0  = (const float*)d_in[7];
    const float* Wrel0 = (const float*)d_in[8];
    const float* Frel0 = (const float*)d_in[9];
    const float* Wsk1  = (const float*)d_in[10];
    const float* Fsk1  = (const float*)d_in[11];
    const float* Wrel1 = (const float*)d_in[12];
    const float* Frel1 = (const float*)d_in[13];
    const float* bng   = (const float*)d_in[14];
    const float* bnb   = (const float*)d_in[15];
    const float* linW  = (const float*)d_in[16];
    const float* linb  = (const float*)d_in[17];
    const float* clfW  = (const float*)d_in[18];
    const float* clfb  = (const float*)d_in[19];

    const int INF = 128, H = 64, R = 4, G = 128;
    const int Nn = in_sizes[0] / INF;       // 100000
    const int E  = in_sizes[1] / 2;         // 1600000
    const int RT = (Nn + 15) / 16;          // row tiles (6250)
    const int* srcP = eidx;
    const int* dstP = eidx + E;

    // ---- workspace carve-up (256B aligned) ----
    char* w = (char*)d_ws;
    size_t off = 0;
    auto carve = [&](size_t bytes) -> void* {
        void* p = (void*)(w + off);
        off = (off + bytes + 255) & ~(size_t)255;
        return p;
    };
    __bf16* Apack = (__bf16*)carve((size_t)RT * 16 * INF * sizeof(__bf16));
    __bf16* Benc  = (__bf16*)carve((size_t)INF * H * sizeof(__bf16));
    __bf16* Bc0   = (__bf16*)carve((size_t)H * 960 * sizeof(__bf16));
    __bf16* Bc1   = (__bf16*)carve((size_t)H * 960 * sizeof(__bf16));
    float*  Ybuf  = (float*)carve((size_t)RT * 16 * 960 * sizeof(float));
    float*  Ha    = (float*)carve((size_t)RT * 16 * H * sizeof(float));
    float*  Hb    = (float*)carve((size_t)RT * 16 * H * sizeof(float));
    float*  Agg   = (float*)carve((size_t)R * Nn * H * sizeof(float));
    float*  Cnt   = (float*)carve((size_t)R * Nn * sizeof(float));
    float*  Gp    = (float*)carve((size_t)G * H * sizeof(float));
    (void)ws_size; (void)n_in; (void)out_size;

    const int B256 = 256;
    const size_t agg4 = ((size_t)R * Nn * H) / 4;
    const size_t cnt4 = (((size_t)R * Nn) + 3) / 4;
    const int gemmGX = (RT + RM - 1) / RM;

    // Edge counts (shared by both layers) + pool accumulator: zero + count.
    zero4_kernel<<<(int)((cnt4 + 255) / 256), B256, 0, stream>>>((float4*)Cnt, cnt4);
    zero4_kernel<<<(G * H / 4 + 255) / 256, B256, 0, stream>>>((float4*)Gp, (size_t)G * H / 4);
    edge_count_kernel<<<(E + 255) / 256, B256, 0, stream>>>(dstP, etype, Cnt, E, Nn);

    // Weight swizzles (tiny).
    pack_b_dense<<<2, B256, 0, stream>>>(encW, Benc, INF, H);            // 512 threads
    pack_b_film<<<15, B256, 0, stream>>>(Wsk0, Fsk0, Wrel0, Frel0, Bc0); // 3840
    pack_b_film<<<15, B256, 0, stream>>>(Wsk1, Fsk1, Wrel1, Frel1, Bc1);

    // Encoder: Ha = x @ enc_W + enc_b
    {
        int tiles = RT * (INF / 32);
        pack_a_kernel<<<(tiles * 32 + 255) / 256, B256, 0, stream>>>(x, Apack, Nn, INF);
        dim3 grid(gemmGX, H / 64);
        gemm_wmma<<<grid, 32, 0, stream>>>(Apack, Benc, Ha, encB, Nn, INF / 32, H, RT);
    }

    // FiLM layer 0: Ha -> Hb  (with BN+ReLU epilogue)
    {
        int tiles = RT * (H / 32);
        pack_a_kernel<<<(tiles * 32 + 255) / 256, B256, 0, stream>>>(Ha, Apack, Nn, H);
        dim3 grid(gemmGX, 960 / 64);
        gemm_wmma<<<grid, 32, 0, stream>>>(Apack, Bc0, Ybuf, nullptr, Nn, H / 32, 960, RT);
        zero4_kernel<<<(int)((agg4 + 255) / 256), B256, 0, stream>>>((float4*)Agg, agg4);
        long long eth = (long long)E * 16;
        edge_msg_kernel<<<(int)((eth + 255) / 256), B256, 0, stream>>>(Ybuf, srcP, dstP, etype, Agg, E, Nn);
        finalize_kernel<<<(Nn * 16 + 255) / 256, B256, 0, stream>>>(Ybuf, Agg, Cnt, bng, bnb, 1, Hb, Nn);
    }

    // FiLM layer 1: Hb -> Ha  (no BN)
    {
        int tiles = RT * (H / 32);
        pack_a_kernel<<<(tiles * 32 + 255) / 256, B256, 0, stream>>>(Hb, Apack, Nn, H);
        dim3 grid(gemmGX, 960 / 64);
        gemm_wmma<<<grid, 32, 0, stream>>>(Apack, Bc1, Ybuf, nullptr, Nn, H / 32, 960, RT);
        zero4_kernel<<<(int)((agg4 + 255) / 256), B256, 0, stream>>>((float4*)Agg, agg4);
        long long eth = (long long)E * 16;
        edge_msg_kernel<<<(int)((eth + 255) / 256), B256, 0, stream>>>(Ybuf, srcP, dstP, etype, Agg, E, Nn);
        finalize_kernel<<<(Nn * 16 + 255) / 256, B256, 0, stream>>>(Ybuf, Agg, Cnt, bng, bnb, 0, Ha, Nn);
    }

    // Global add-pool + MLP head.
    pool_kernel<<<(Nn * 16 + 255) / 256, B256, 0, stream>>>(Ha, batch, Gp, Nn);
    head_kernel<<<1, 128, 0, stream>>>(Gp, linW, linb, clfW, clfb, (float*)d_out);
}
